// MLCV_ProposalModule_47828755808284
// MI455X (gfx1250) — matrix operand order repaired and checked by reading, not compiled, not run on hardware
//
#include <hip/hip_runtime.h>

typedef __attribute__((ext_vector_type(16))) _Float16 v16h;
typedef __attribute__((ext_vector_type(8)))  float    v8f;

#define B_    32
#define KPTS  1024
#define NP    256
#define NS    16
#define SEEDC 256
#define CIN   259          // 3 + 256
#define KPAD0 288          // 259 padded to multiple of 32
#define J2D   (B_*NP*NS)   // 131072 grouped columns
#define J1D   (B_*NP)      // 8192 proposal columns
#define OUTD  119
#define EPSV  1e-5f

// ---------------- output slab offsets (f32 elements) ----------------
#define O_OBJ 0
#define O_CEN 16384
#define O_HS  40960
#define O_HR  139264
#define O_SS  237568
#define O_SR  385024
#define O_SEM 827392

// =====================================================================
// FPS: one block per batch, 256 iterations over 1024 points
// =====================================================================
__global__ void fps_kernel(const float* __restrict__ xyz, float* __restrict__ new_xyz) {
  __shared__ float mind[KPTS];
  __shared__ float red[256];
  __shared__ int   redi[256];
  __shared__ int   sfar;
  int b = blockIdx.x, t = threadIdx.x;
  const float* xb = xyz + (size_t)b * KPTS * 3;
  for (int i = t; i < KPTS; i += 256) mind[i] = 1e10f;
  if (t == 0) sfar = 0;
  __syncthreads();
  for (int it = 0; it < NP; ++it) {
    int far = sfar;
    if (t == 0) {
      new_xyz[((size_t)b * NP + it) * 3 + 0] = xb[far * 3 + 0];
      new_xyz[((size_t)b * NP + it) * 3 + 1] = xb[far * 3 + 1];
      new_xyz[((size_t)b * NP + it) * 3 + 2] = xb[far * 3 + 2];
    }
    float cx = xb[far * 3 + 0], cy = xb[far * 3 + 1], cz = xb[far * 3 + 2];
    float best = -1.0f; int bi = 0;
    for (int i = t; i < KPTS; i += 256) {
      float dx = xb[i * 3 + 0] - cx, dy = xb[i * 3 + 1] - cy, dz = xb[i * 3 + 2] - cz;
      float d = dx * dx + dy * dy + dz * dz;
      float m = fminf(mind[i], d);
      mind[i] = m;
      if (m > best) { best = m; bi = i; }
    }
    red[t] = best; redi[t] = bi;
    __syncthreads();
    for (int s = 128; s > 0; s >>= 1) {
      if (t < s) {
        if (red[t + s] > red[t] || (red[t + s] == red[t] && redi[t + s] < redi[t])) {
          red[t] = red[t + s]; redi[t] = redi[t + s];
        }
      }
      __syncthreads();
    }
    if (t == 0) sfar = redi[0];
    __syncthreads();
  }
}

// =====================================================================
// Ball query: one thread per (b, proposal)
// =====================================================================
__global__ void ballq_kernel(const float* __restrict__ xyz,
                             const float* __restrict__ new_xyz,
                             int* __restrict__ idx) {
  int tid = blockIdx.x * blockDim.x + threadIdx.x;
  if (tid >= B_ * NP) return;
  int b = tid / NP;
  const float* xb = xyz + (size_t)b * KPTS * 3;
  float cx = new_xyz[tid * 3 + 0], cy = new_xyz[tid * 3 + 1], cz = new_xyz[tid * 3 + 2];
  int out[NS]; int found = 0; int first = 0; bool has = false;
  for (int k = 0; k < KPTS && found < NS; ++k) {
    float dx = xb[k * 3 + 0] - cx, dy = xb[k * 3 + 1] - cy, dz = xb[k * 3 + 2] - cz;
    if (dx * dx + dy * dy + dz * dz < 0.09f) {
      if (!has) { has = true; first = k; }
      out[found++] = k;
    }
  }
  for (int i = found; i < NS; ++i) out[i] = first;
  for (int i = 0; i < NS; ++i) idx[(size_t)tid * NS + i] = out[i];
}

// =====================================================================
// Gather + normalize xyz + pack f16 activation rows [J2D][KPAD0]
// =====================================================================
__global__ void gather_kernel(const float* __restrict__ xyz,
                              const float* __restrict__ features,
                              const float* __restrict__ new_xyz,
                              const int* __restrict__ idx,
                              _Float16* __restrict__ X) {
  int j = blockIdx.x;      // column
  int c = threadIdx.x;     // channel 0..287
  int bs = j / NS;
  int b = bs / NP;
  int pt = idx[j];
  float v;
  if (c < 3)        v = (xyz[((size_t)b * KPTS + pt) * 3 + c] - new_xyz[(size_t)bs * 3 + c]) / 0.3f;
  else if (c < CIN) v = features[((size_t)b * SEEDC + (c - 3)) * KPTS + pt];
  else              v = 0.0f;
  X[(size_t)j * KPAD0 + c] = (_Float16)v;
}

// =====================================================================
// Weights f32 [128][Kin] -> fragment-major f16 A buffer:
//   Aswz[kc][rt][lane][e]  (kc = K/32 chunk, rt = 16-row tile, lane 0..31,
//   e 0..15 holds k = 16*(lane/16) + e in the chunk, the order the GEMM loads X)
// GEMM then loads each fragment as 32 contiguous bytes per lane.
// =====================================================================
__global__ void convw_swz_kernel(const float* __restrict__ w, _Float16* __restrict__ out,
                                 int Kin, int Kpad) {
  int t = blockIdx.x * blockDim.x + threadIdx.x;
  if (t >= 128 * Kpad) return;
  int e    = t & 15;
  int lane = (t >> 4) & 31;
  int rt   = (t >> 9) & 7;
  int kc   = t >> 12;
  int half = lane >> 4, l16 = lane & 15;
  int m = rt * 16 + l16;
  int v = e >> 1, p = e & 1;
  int koff = half * 16 + (v << 1) + p;
  int k = kc * 32 + koff;
  out[t] = (_Float16)(k < Kin ? w[m * Kin + k] : 0.0f);
}

// =====================================================================
// WMMA GEMM: Y[j][o] = sum_k A[o][k] * X[j][k],  O = 128 fixed.
// One wave = TWO 16-column tiles x all 8 row tiles; A fragment loaded once
// per (kc, rt) and fed to both column tiles. __launch_bounds__(256,1)
// pins low occupancy so the ~160 live VGPRs (128 accum) never spill.
// =====================================================================
__global__ __launch_bounds__(256, 1)
void wmma_gemm128(const _Float16* __restrict__ Aswz,
                  const _Float16* __restrict__ X,
                  float* __restrict__ Y, int Kpad, int J) {
  int lane = threadIdx.x & 31;
  int wav  = blockIdx.x * (blockDim.x >> 5) + (threadIdx.x >> 5);
  int j0 = wav * 32;
  if (j0 >= J) return;
  int l16 = lane & 15, half = lane >> 4;
  v8f acc0[8], acc1[8];
#pragma unroll
  for (int r = 0; r < 8; ++r) {
    acc0[r] = (v8f){0, 0, 0, 0, 0, 0, 0, 0};
    acc1[r] = (v8f){0, 0, 0, 0, 0, 0, 0, 0};
  }
  const _Float16* xcol0 = X + (size_t)(j0 + l16) * Kpad;
  const _Float16* xcol1 = X + (size_t)(j0 + 16 + l16) * Kpad;
  const _Float16* ap = Aswz + lane * 16;          // lane's slice of fragment 0
  for (int kc = 0; kc < Kpad; kc += 32) {
    union { v16h v; unsigned u[8]; } b0, b1;
    const unsigned* x0 = (const unsigned*)(xcol0 + kc + half * 16);
    const unsigned* x1 = (const unsigned*)(xcol1 + kc + half * 16);
#pragma unroll
    for (int v = 0; v < 8; ++v) { b0.u[v] = x0[v]; b1.u[v] = x1[v]; }
    if (kc + 32 < Kpad) {                         // prefetch next K chunk of X
      __builtin_prefetch(xcol0 + kc + 32 + half * 16, 0, 1);
      __builtin_prefetch(xcol1 + kc + 32 + half * 16, 0, 1);
    }
    const _Float16* abase = ap + (size_t)(kc >> 5) * (8 * 32 * 16);
#pragma unroll
    for (int rt = 0; rt < 8; ++rt) {
      union { v16h v; unsigned u[8]; } af;
      const unsigned* as = (const unsigned*)(abase + rt * (32 * 16));
#pragma unroll
      for (int v = 0; v < 8; ++v) af.u[v] = as[v];
      acc0[rt] = __builtin_amdgcn_wmma_f32_16x16x32_f16(
          false, af.v, false, b0.v, (short)0, acc0[rt], false, false);
      acc1[rt] = __builtin_amdgcn_wmma_f32_16x16x32_f16(
          false, af.v, false, b1.v, (short)0, acc1[rt], false, false);
    }
  }
#pragma unroll
  for (int rt = 0; rt < 8; ++rt) {
    float* y0 = Y + (size_t)(j0 + l16) * 128 + rt * 16 + half * 8;
    float* y1 = Y + (size_t)(j0 + 16 + l16) * 128 + rt * 16 + half * 8;
#pragma unroll
    for (int r = 0; r < 8; ++r) { y0[r] = acc0[rt][r]; y1[r] = acc1[rt][r]; }
  }
}

// =====================================================================
// Cross-batch BN: deterministic two-stage per-channel stats over [J][128]
// =====================================================================
__global__ void bn_partial_kernel(const float* __restrict__ Y, float* __restrict__ psum,
                                  float* __restrict__ psq, int rowsPerBlock, int J) {
  int o = threadIdx.x;                 // 128 channels
  int p = blockIdx.x;
  int r0 = p * rowsPerBlock;
  int r1 = r0 + rowsPerBlock; if (r1 > J) r1 = J;
  float s = 0.f, ss = 0.f;
  for (int r = r0; r < r1; ++r) {
    float v = Y[(size_t)r * 128 + o];
    s += v; ss += v * v;
  }
  psum[(size_t)p * 128 + o] = s;
  psq [(size_t)p * 128 + o] = ss;
}

__global__ void bn_reduce_kernel(const float* __restrict__ psum, const float* __restrict__ psq,
                                 int nP, float invJ, float* __restrict__ mean,
                                 float* __restrict__ rstd) {
  int o = threadIdx.x;
  float s = 0.f, ss = 0.f;
  for (int p = 0; p < nP; ++p) { s += psum[(size_t)p * 128 + o]; ss += psq[(size_t)p * 128 + o]; }
  float m = s * invJ;
  float v = ss * invJ - m * m;
  mean[o] = m;
  rstd[o] = rsqrtf(v + EPSV);
}

__global__ void bn_norm_kernel(const float* __restrict__ Y, const float* __restrict__ mean,
                               const float* __restrict__ rstd, const float* __restrict__ g,
                               const float* __restrict__ bta, long N,
                               _Float16* __restrict__ out16, float* __restrict__ out32) {
  long t = (long)blockIdx.x * blockDim.x + threadIdx.x;
  if (t >= N) return;
  int o = (int)(t & 127);
  float v = (Y[t] - mean[o]) * rstd[o] * g[o] + bta[o];
  v = fmaxf(v, 0.0f);
  if (out16) out16[t] = (_Float16)v;
  if (out32) out32[t] = v;
}

// max over NS samples: [J2D][128] -> feat [B][128][NP]
__global__ void maxpool_kernel(const float* __restrict__ Yn, float* __restrict__ feat) {
  int t = blockIdx.x * blockDim.x + threadIdx.x;
  if (t >= B_ * NP * 128) return;
  int o = t & 127;
  int bs = t >> 7;
  int b = bs / NP, s = bs % NP;
  float m = -1e30f;
  for (int n = 0; n < NS; ++n) m = fmaxf(m, Yn[((size_t)bs * NS + n) * 128 + o]);
  feat[((size_t)b * 128 + o) * NP + s] = m;
}

// per-(b,c) max over contiguous L (gf1 from seed_features, gf2 from feat)
__global__ void rowmax_kernel(const float* __restrict__ src, float* __restrict__ dst,
                              int C, int L) {
  int t = blockIdx.x * blockDim.x + threadIdx.x;
  if (t >= B_ * C) return;
  const float* p = src + (size_t)t * L;
  float m = -1e30f;
  for (int i = 0; i < L; ++i) m = fmaxf(m, p[i]);
  dst[t] = m;
}

// =====================================================================
// CGNL block kernels (x layout [b][128][NP])
// =====================================================================
__global__ void tpg_kernel(const float* __restrict__ x, const float* __restrict__ wt,
                           const float* __restrict__ wp, const float* __restrict__ wg,
                           float* __restrict__ tpg) {
  int t = blockIdx.x * blockDim.x + threadIdx.x;
  if (t >= B_ * 3 * 64 * NP) return;
  int n   = t % NP;
  int pch = (t / NP) % 64;
  int wch = (t / (NP * 64)) % 3;
  int b   = t / (NP * 64 * 3);
  const float* w  = (wch == 0) ? wt : ((wch == 1) ? wp : wg);
  const float* xb = x + (size_t)b * 128 * NP;
  float s = 0.f;
  for (int c = 0; c < 128; ++c) s += w[pch * 128 + c] * xb[(size_t)c * NP + n];
  tpg[t] = s;
}

__global__ void att_kernel(const float* __restrict__ tpg, float* __restrict__ att) {
  int bg = blockIdx.x;            // b*4+g
  int b = bg >> 2, g = bg & 3;
  const float* pb = tpg + ((size_t)(b * 3 + 1) * 64 + g * 16) * NP;
  const float* gb = tpg + ((size_t)(b * 3 + 2) * 64 + g * 16) * NP;
  float s = 0.f;
  for (int i = threadIdx.x; i < 16 * NP; i += blockDim.x) s += pb[i] * gb[i];
  __shared__ float red[256];
  red[threadIdx.x] = s; __syncthreads();
  for (int st = 128; st > 0; st >>= 1) {
    if (threadIdx.x < st) red[threadIdx.x] += red[threadIdx.x + st];
    __syncthreads();
  }
  if (threadIdx.x == 0) att[bg] = red[0];
}

__global__ void zmix_kernel(const float* __restrict__ tpg, const float* __restrict__ att,
                            const float* __restrict__ wz, float* __restrict__ z) {
  int t = blockIdx.x * blockDim.x + threadIdx.x;
  if (t >= B_ * 128 * NP) return;
  int n = t % NP; int ch = (t / NP) & 127; int b = t / (NP * 128);
  int g = ch >> 5, o = ch & 31;
  const float* tb = tpg + ((size_t)(b * 3 + 0) * 64 + g * 16) * NP;
  float s = 0.f;
  for (int c = 0; c < 16; ++c) s += wz[(g * 32 + o) * 16 + c] * tb[(size_t)c * NP + n];
  z[t] = att[b * 4 + g] * s;
}

__global__ void gnstat_kernel(const float* __restrict__ z, float* __restrict__ gm,
                              float* __restrict__ gr) {
  int bg = blockIdx.x; int b = bg >> 2, g = bg & 3;
  const float* zp = z + ((size_t)b * 128 + g * 32) * NP;
  float s = 0.f, ss = 0.f;
  for (int i = threadIdx.x; i < 32 * NP; i += blockDim.x) { float v = zp[i]; s += v; ss += v * v; }
  __shared__ float r1[256], r2[256];
  r1[threadIdx.x] = s; r2[threadIdx.x] = ss; __syncthreads();
  for (int st = 128; st > 0; st >>= 1) {
    if (threadIdx.x < st) { r1[threadIdx.x] += r1[threadIdx.x + st]; r2[threadIdx.x] += r2[threadIdx.x + st]; }
    __syncthreads();
  }
  if (threadIdx.x == 0) {
    float m = r1[0] / 8192.0f;
    float v = r2[0] / 8192.0f - m * m;
    gm[bg] = m; gr[bg] = rsqrtf(v + EPSV);
  }
}

__global__ void gnout_kernel(const float* __restrict__ z, const float* __restrict__ gm,
                             const float* __restrict__ gr, const float* __restrict__ gng,
                             const float* __restrict__ gnb, const float* __restrict__ x,
                             float* __restrict__ out) {
  int t = blockIdx.x * blockDim.x + threadIdx.x;
  if (t >= B_ * 128 * NP) return;
  int ch = (t / NP) & 127; int b = t / (NP * 128);
  int g = ch >> 5;
  out[t] = (z[t] - gm[b * 4 + g]) * gr[b * 4 + g] * gng[ch] + gnb[ch] + x[t];
}

// =====================================================================
// Gating path
// =====================================================================
__global__ void partA_kernel(const float* __restrict__ gsw, const float* __restrict__ gf1,
                             const float* __restrict__ gf2, float* __restrict__ pA) {
  int t = blockIdx.x * blockDim.x + threadIdx.x;
  if (t >= B_ * 128) return;
  int o = t & 127, b = t >> 7;
  float s = 0.f;
  for (int c = 0; c < 256; ++c) s += gsw[o * 512 + c] * gf1[b * 256 + c];
  for (int c = 0; c < 128; ++c) s += gsw[o * 512 + 256 + c] * gf2[b * 128 + c];
  pA[t] = s;
}

// gate = sigmoid(log|g|) = |g|/(1+|g|); emit gated net as f16 [J1D][128]
__global__ void gatepack_kernel(const float* __restrict__ net, const float* __restrict__ gsw,
                                const float* __restrict__ gsb, const float* __restrict__ pA,
                                _Float16* __restrict__ netbf) {
  int t = blockIdx.x * blockDim.x + threadIdx.x;
  if (t >= B_ * NP * 128) return;
  int o = t & 127; int bn = t >> 7;
  int b = bn / NP, n = bn % NP;
  const float* nb = net + (size_t)b * 128 * NP;
  float gate = pA[b * 128 + o] + gsb[o];
  for (int c = 0; c < 128; ++c) gate += gsw[o * 512 + 384 + c] * nb[(size_t)c * NP + n];
  float a = fabsf(gate);
  gate = a / (1.0f + a);
  netbf[t] = (_Float16)(nb[(size_t)o * NP + n] * gate);
}

// final head + output scatter (reads net2f [J1D][128] f32)
__global__ void c3out_kernel(const float* __restrict__ net2f, const float* __restrict__ w,
                             const float* __restrict__ bias, const float* __restrict__ new_xyz,
                             float* __restrict__ out) {
  int t = blockIdx.x * blockDim.x + threadIdx.x;
  if (t >= B_ * NP * 128) return;
  int o = t & 127; if (o >= OUTD) return;
  int bn = t >> 7;
  const float* xp = net2f + (size_t)bn * 128;
  float s = bias[o];
  for (int c = 0; c < 128; ++c) s += w[o * 128 + c] * xp[c];
  if (o < 2)        out[O_OBJ + (size_t)bn * 2 + o] = s;
  else if (o < 5)   out[O_CEN + (size_t)bn * 3 + (o - 2)] = new_xyz[(size_t)bn * 3 + (o - 2)] + s;
  else if (o < 17)  out[O_HS  + (size_t)bn * 12 + (o - 5)] = s;
  else if (o < 29)  out[O_HR  + (size_t)bn * 12 + (o - 17)] = s;
  else if (o < 47)  out[O_SS  + (size_t)bn * 18 + (o - 29)] = s;
  else if (o < 101) out[O_SR  + (size_t)bn * 54 + (o - 47)] = s;
  else              out[O_SEM + (size_t)bn * 18 + (o - 101)] = s;
}

// =====================================================================
// Host launch
// =====================================================================
static inline unsigned cdiv(long a, long b) { return (unsigned)((a + b - 1) / b); }

extern "C" void kernel_launch(void* const* d_in, const int* in_sizes, int n_in,
                              void* d_out, int out_size, void* d_ws, size_t ws_size,
                              hipStream_t stream) {
  (void)in_sizes; (void)n_in; (void)out_size; (void)ws_size;
  const float* xyz   = (const float*)d_in[0];
  const float* feats = (const float*)d_in[1];
  const float* seed  = (const float*)d_in[2];
  const float* mlp_w0 = (const float*)d_in[3];
  const float* mlp_w1 = (const float*)d_in[4];
  const float* mlp_w2 = (const float*)d_in[5];
  const float* mlp_g[3] = {(const float*)d_in[6], (const float*)d_in[8], (const float*)d_in[10]};
  const float* mlp_b[3] = {(const float*)d_in[7], (const float*)d_in[9], (const float*)d_in[11]};
  const float* sa_t[2]   = {(const float*)d_in[12], (const float*)d_in[18]};
  const float* sa_p[2]   = {(const float*)d_in[13], (const float*)d_in[19]};
  const float* sa_g[2]   = {(const float*)d_in[14], (const float*)d_in[20]};
  const float* sa_z[2]   = {(const float*)d_in[15], (const float*)d_in[21]};
  const float* sa_gng[2] = {(const float*)d_in[16], (const float*)d_in[22]};
  const float* sa_gnb[2] = {(const float*)d_in[17], (const float*)d_in[23]};
  const float* gs_w = (const float*)d_in[24];
  const float* gs_b = (const float*)d_in[25];
  const float* c1_w = (const float*)d_in[26];
  const float* c2_w = (const float*)d_in[28];
  const float* c3_w = (const float*)d_in[30];
  const float* c3_b = (const float*)d_in[31];
  const float* bn1_g = (const float*)d_in[32];
  const float* bn1_b = (const float*)d_in[33];
  const float* bn2_g = (const float*)d_in[34];
  const float* bn2_b = (const float*)d_in[35];
  float* out = (float*)d_out;

  char* ws = (char*)d_ws;
  // ---- workspace layout ----
  float*  new_xyz = (float*)(ws + 32768);                 // 24576 f32
  int*    idx     = (int*)(ws + 131072);                  // 131072 i32
  float*  bnMean  = (float*)(ws + 655360);                // 128
  float*  bnRstd  = (float*)(ws + 655360 + 512);          // 128
  float*  psum    = (float*)(ws + 655360 + 4096);         // up to 512*128
  float*  psq     = (float*)(ws + 655360 + 4096 + 262144);
  _Float16* W0h  = (_Float16*)(ws + 1703936);             // swizzled 128*288
  _Float16* W1h  = W0h + 128 * KPAD0;
  _Float16* W2h  = W1h + 128 * 128;
  _Float16* Wc1h = W2h + 128 * 128;
  _Float16* Wc2h = Wc1h + 128 * 128;
  float* attb  = (float*)(ws + 2097152);                  // 128
  float* gnm   = (float*)(ws + 2097152 + 1024);           // 128
  float* gnr   = (float*)(ws + 2097152 + 1536);           // 128
  float* gf1   = (float*)(ws + 2097152 + 65536);          // 8192
  float* gf2   = (float*)(ws + 2097152 + 131072);         // 4096
  float* pA    = (float*)(ws + 2097152 + 196608);         // 4096
  char*  XB    = ws + 8388608;                            // 75,497,472 B region
  _Float16* X0 = (_Float16*)XB;                           // [J2D][288]
  float*  YB   = (float*)(ws + 83886080);                 // [J2D][128] f32
  _Float16* X1 = (_Float16*)(ws + 150994944);             // [J2D][128] f16
  _Float16* X2 = (_Float16*)XB;                           // reuse as [J2D][128] f16
  // post-pool buffers reuse XB region (X consumed by then)
  float* feat  = (float*)(XB + 0);                        // [B][128][NP]
  float* netA  = (float*)(XB + 4194304);
  float* netB  = (float*)(XB + 8388608);
  float* tpg   = (float*)(XB + 12582912);                 // [B][3][64][NP]
  float* zbuf  = (float*)(XB + 18874368);                 // [B][128][NP]
  _Float16* netbf  = (_Float16*)(XB + 23068672);          // [J1D][128]
  _Float16* netbf2 = (_Float16*)(XB + 25165824);
  float* net2f = (float*)(XB + 27262976);                 // [J1D][128]

  // ---- sampling / grouping ----
  fps_kernel<<<B_, 256, 0, stream>>>(xyz, new_xyz);
  ballq_kernel<<<cdiv(B_ * NP, 256), 256, 0, stream>>>(xyz, new_xyz, idx);
  gather_kernel<<<J2D, KPAD0, 0, stream>>>(xyz, feats, new_xyz, idx, X0);

  // ---- weight conversion (fragment-major swizzle) ----
  convw_swz_kernel<<<cdiv(128 * KPAD0, 256), 256, 0, stream>>>(mlp_w0, W0h, CIN, KPAD0);
  convw_swz_kernel<<<cdiv(128 * 128, 256), 256, 0, stream>>>(mlp_w1, W1h, 128, 128);
  convw_swz_kernel<<<cdiv(128 * 128, 256), 256, 0, stream>>>(mlp_w2, W2h, 128, 128);
  convw_swz_kernel<<<cdiv(128 * 128, 256), 256, 0, stream>>>(c1_w, Wc1h, 128, 128);
  convw_swz_kernel<<<cdiv(128 * 128, 256), 256, 0, stream>>>(c2_w, Wc2h, 128, 128);

  const long N2 = (long)J2D * 128;
  // ---- conv layer 0 ----
  wmma_gemm128<<<J2D / 32 / 8, 256, 0, stream>>>(W0h, X0, YB, KPAD0, J2D);
  bn_partial_kernel<<<512, 128, 0, stream>>>(YB, psum, psq, 256, J2D);
  bn_reduce_kernel<<<1, 128, 0, stream>>>(psum, psq, 512, 1.0f / J2D, bnMean, bnRstd);
  bn_norm_kernel<<<cdiv(N2, 256), 256, 0, stream>>>(YB, bnMean, bnRstd, mlp_g[0], mlp_b[0], N2, X1, nullptr);
  // ---- conv layer 1 ----
  wmma_gemm128<<<J2D / 32 / 8, 256, 0, stream>>>(W1h, X1, YB, 128, J2D);
  bn_partial_kernel<<<512, 128, 0, stream>>>(YB, psum, psq, 256, J2D);
  bn_reduce_kernel<<<1, 128, 0, stream>>>(psum, psq, 512, 1.0f / J2D, bnMean, bnRstd);
  bn_norm_kernel<<<cdiv(N2, 256), 256, 0, stream>>>(YB, bnMean, bnRstd, mlp_g[1], mlp_b[1], N2, X2, nullptr);
  // ---- conv layer 2 (normalize in place, f32) ----
  wmma_gemm128<<<J2D / 32 / 8, 256, 0, stream>>>(W2h, X2, YB, 128, J2D);
  bn_partial_kernel<<<512, 128, 0, stream>>>(YB, psum, psq, 256, J2D);
  bn_reduce_kernel<<<1, 128, 0, stream>>>(psum, psq, 512, 1.0f / J2D, bnMean, bnRstd);
  bn_norm_kernel<<<cdiv(N2, 256), 256, 0, stream>>>(YB, bnMean, bnRstd, mlp_g[2], mlp_b[2], N2, nullptr, YB);

  // ---- max-pool over samples ----
  maxpool_kernel<<<cdiv(B_ * NP * 128, 256), 256, 0, stream>>>(YB, feat);
  rowmax_kernel<<<cdiv(B_ * 256, 256), 256, 0, stream>>>(seed, gf1, 256, KPTS);
  rowmax_kernel<<<cdiv(B_ * 128, 256), 256, 0, stream>>>(feat, gf2, 128, NP);

  // ---- two CGNL blocks ----
  const float* cg_in[2]  = {feat, netA};
  float*       cg_out[2] = {netA, netB};
  for (int s = 0; s < 2; ++s) {
    tpg_kernel<<<cdiv(B_ * 3 * 64 * NP, 256), 256, 0, stream>>>(cg_in[s], sa_t[s], sa_p[s], sa_g[s], tpg);
    att_kernel<<<B_ * 4, 256, 0, stream>>>(tpg, attb);
    zmix_kernel<<<cdiv(B_ * 128 * NP, 256), 256, 0, stream>>>(tpg, attb, sa_z[s], zbuf);
    gnstat_kernel<<<B_ * 4, 256, 0, stream>>>(zbuf, gnm, gnr);
    gnout_kernel<<<cdiv(B_ * 128 * NP, 256), 256, 0, stream>>>(zbuf, gnm, gnr, sa_gng[s], sa_gnb[s], cg_in[s], cg_out[s]);
  }

  // ---- gating + pack ----
  partA_kernel<<<cdiv(B_ * 128, 256), 256, 0, stream>>>(gs_w, gf1, gf2, pA);
  gatepack_kernel<<<cdiv(B_ * NP * 128, 256), 256, 0, stream>>>(netB, gs_w, gs_b, pA, netbf);

  const long N1 = (long)J1D * 128;
  // ---- conv1d c1 (bias cancels inside BN) ----
  wmma_gemm128<<<J1D / 32 / 8, 256, 0, stream>>>(Wc1h, netbf, YB, 128, J1D);
  bn_partial_kernel<<<32, 128, 0, stream>>>(YB, psum, psq, 256, J1D);
  bn_reduce_kernel<<<1, 128, 0, stream>>>(psum, psq, 32, 1.0f / J1D, bnMean, bnRstd);
  bn_norm_kernel<<<cdiv(N1, 256), 256, 0, stream>>>(YB, bnMean, bnRstd, bn1_g, bn1_b, N1, netbf2, nullptr);
  // ---- conv1d c2 ----
  wmma_gemm128<<<J1D / 32 / 8, 256, 0, stream>>>(Wc2h, netbf2, YB, 128, J1D);
  bn_partial_kernel<<<32, 128, 0, stream>>>(YB, psum, psq, 256, J1D);
  bn_reduce_kernel<<<1, 128, 0, stream>>>(psum, psq, 32, 1.0f / J1D, bnMean, bnRstd);
  bn_norm_kernel<<<cdiv(N1, 256), 256, 0, stream>>>(YB, bnMean, bnRstd, bn2_g, bn2_b, N1, nullptr, net2f);

  // ---- head + scatter to outputs ----
  c3out_kernel<<<cdiv(B_ * NP * 128, 256), 256, 0, stream>>>(net2f, c3_w, c3_b, new_xyz, out);
}
